// GATModel_TRANS_56581899158035
// MI455X (gfx1250) — matrix operand order repaired and checked by reading, not compiled, not run on hardware
//
#include <hip/hip_runtime.h>
#include <math.h>

// ---------------- problem constants (from reference) ----------------
constexpr int NN  = 10000;
constexpr int EE  = 30000;
constexpr int IND = 1024;
constexpr int H1c = 4;
constexpr int C1c = 512;
constexpr int D1c = H1c * C1c;      // 2048
constexpr int H2c = 8;
constexpr int C2c = 460;
constexpr int D2c = H2c * C2c;      // 3680
constexpr int E2c = EE + NN;        // 40000 (edges + self loops)

// ---------------- WMMA types ----------------
typedef __attribute__((ext_vector_type(16))) _Float16 v16h;
typedef __attribute__((ext_vector_type(8)))  float    v8f;

// =====================================================================
// Tiled WMMA GEMM:  C[M,Nt] = A[M,K] (fp32) * B[K,Nt] (fp32) + bias[Nt]
// fp32 -> f16 on the fly, f32 accumulate.
// Block = 256 threads (8 waves), macro-tile 128x128, K-step 32,
// double-buffered LDS, register prefetch, one barrier per K-step.
// Each wave: 32x64 strip, 8 accumulators, 8 WMMAs per K-step.
// =====================================================================
__global__ __launch_bounds__(256) void gemm_wmma_f16(
    const float* __restrict__ A, const float* __restrict__ B,
    const float* __restrict__ bias, float* __restrict__ C,
    int M, int K, int Nt)
{
    __shared__ _Float16 As[2][128][36];   // [buf][m][k], padded (8B-aligned rows)
    __shared__ _Float16 Bs[2][128][36];   // [buf][n][k], B transposed

    const int t    = threadIdx.x;
    const int lane = t & 31;
    const int wave = t >> 5;
    const int rowBase = blockIdx.y * 128;
    const int colBase = blockIdx.x * 128;

    // wave -> 32-row x 64-col strip of the 128x128 tile
    const int wr = (wave >> 1) * 32;      // 0,32,64,96
    const int wc = (wave & 1) * 64;       // 0,64

    const int nIdx = lane & 15;
    const int krA  = (lane < 16) ? 0 : 16;      // A K-half     (contiguous, matches B)
    const int kbB  = (lane < 16) ? 0 : 16;      // B K-half     (contiguous, matches A)

    v8f zero = {0.f,0.f,0.f,0.f,0.f,0.f,0.f,0.f};
    v8f acc[2][4] = {{zero, zero, zero, zero}, {zero, zero, zero, zero}};

    // ---- tile loader mappings ----
    // A: thread t covers rows arow+32i, k = ac4..ac4+3 (contiguous, coalesced)
    const int arow = t >> 3;              // 0..31 (+ i*32)
    const int ac4  = (t & 7) * 4;         // 0..28
    // B: thread t covers k = bk4..bk4+3 (consecutive rows), n = bn0..bn0+3
    //    -> per-n LDS writes are 4 consecutive f16 (merge to ds_store_b64)
    const int bk4  = (t >> 5) * 4;        // 0,4,...,28
    const int bn0  = (t & 31) * 4;        // 0..124
    const int gcolB = colBase + bn0;

    auto loadA = [&](int kt, float4* ra) {
        const int kbase = kt * 32;
        #pragma unroll
        for (int i = 0; i < 4; ++i) {
            const int grow = rowBase + arow + i * 32;
            ra[i] = (grow < M)
                ? *reinterpret_cast<const float4*>(A + (size_t)grow * K + kbase + ac4)
                : make_float4(0.f, 0.f, 0.f, 0.f);
        }
    };
    auto loadB = [&](int kt, float4* rb) {
        const int kbase = kt * 32;
        #pragma unroll
        for (int i = 0; i < 4; ++i) {
            rb[i] = (gcolB < Nt)     // Nt multiple of 4 -> full float4 in range
                ? *reinterpret_cast<const float4*>(B + (size_t)(kbase + bk4 + i) * Nt + gcolB)
                : make_float4(0.f, 0.f, 0.f, 0.f);
        }
    };
    auto storeTiles = [&](int buf, const float4* ra, const float4* rb) {
        #pragma unroll
        for (int i = 0; i < 4; ++i) {
            const int r = arow + i * 32;
            As[buf][r][ac4 + 0] = (_Float16)ra[i].x;
            As[buf][r][ac4 + 1] = (_Float16)ra[i].y;
            As[buf][r][ac4 + 2] = (_Float16)ra[i].z;
            As[buf][r][ac4 + 3] = (_Float16)ra[i].w;
        }
        // transpose in registers: rb[i] holds K=bk4+i for N=bn0..bn0+3
        const float* rbf = reinterpret_cast<const float*>(rb);  // rbf[i*4 + j]
        #pragma unroll
        for (int j = 0; j < 4; ++j) {
            Bs[buf][bn0 + j][bk4 + 0] = (_Float16)rbf[0 * 4 + j];
            Bs[buf][bn0 + j][bk4 + 1] = (_Float16)rbf[1 * 4 + j];
            Bs[buf][bn0 + j][bk4 + 2] = (_Float16)rbf[2 * 4 + j];
            Bs[buf][bn0 + j][bk4 + 3] = (_Float16)rbf[3 * 4 + j];
        }
    };

    const int numK = K >> 5;              // K multiple of 32 for all calls

    // ---- prologue: fill buffer 0 ----
    float4 ra[4], rb[4];
    loadA(0, ra); loadB(0, rb);
    storeTiles(0, ra, rb);
    __syncthreads();

    for (int kt = 0; kt < numK; ++kt) {
        const int cur = kt & 1;

        // prefetch next K-tile into registers (overlaps WMMA below)
        if (kt + 1 < numK) { loadA(kt + 1, ra); loadB(kt + 1, rb); }
        // pull K+2 tile toward L2 (global_prefetch_b8)
        if (kt + 2 < numK) {
            const int grow = rowBase + arow;
            if (grow < M)
                __builtin_prefetch(A + (size_t)grow * K + (kt + 2) * 32 + ac4, 0, 1);
            if (gcolB < Nt)
                __builtin_prefetch(B + (size_t)((kt + 2) * 32 + bk4) * Nt + gcolB, 0, 1);
        }

        // ---- A fragments (two 16x32 m-tiles) ----
        v16h afrag[2];
        #pragma unroll
        for (int mi = 0; mi < 2; ++mi) {
            const int m = wr + mi * 16 + nIdx;
            #pragma unroll
            for (int j = 0; j < 8; ++j) {
                afrag[mi][j]     = As[cur][m][krA + j];
                afrag[mi][8 + j] = As[cur][m][krA + 8 + j];
            }
        }
        // ---- 4 B fragments, 8 WMMAs ----
        #pragma unroll
        for (int ni = 0; ni < 4; ++ni) {
            const int n = wc + ni * 16 + nIdx;
            v16h bfrag;
            #pragma unroll
            for (int j = 0; j < 16; ++j) bfrag[j] = Bs[cur][n][kbB + j];
            acc[0][ni] = __builtin_amdgcn_wmma_f32_16x16x32_f16(
                false, afrag[0], false, bfrag, (short)0, acc[0][ni], false, false);
            acc[1][ni] = __builtin_amdgcn_wmma_f32_16x16x32_f16(
                false, afrag[1], false, bfrag, (short)0, acc[1][ni], false, false);
        }

        // write next tile into the other buffer; safe: its last readers
        // finished before the barrier that ended the previous iteration
        if (kt + 1 < numK) storeTiles(cur ^ 1, ra, rb);
        __syncthreads();
    }

    // ---- store: C/D layout (lanes 0-15 -> M 0..7, lanes 16-31 -> M 8..15) ----
    const int mC = (lane < 16) ? 0 : 8;
    #pragma unroll
    for (int mi = 0; mi < 2; ++mi) {
        #pragma unroll
        for (int ni = 0; ni < 4; ++ni) {
            const int col = colBase + wc + ni * 16 + nIdx;
            if (col < Nt) {
                const float bv = bias[col];
                #pragma unroll
                for (int i = 0; i < 8; ++i) {
                    const int row = rowBase + wr + mi * 16 + mC + i;
                    if (row < M)
                        C[(size_t)row * Nt + col] = acc[mi][ni][i] + bv;
                }
            }
        }
    }
}

// =====================================================================
// Utility + edge/segment kernels
// =====================================================================
__global__ void fill_kernel(float* __restrict__ p, float val, size_t n) {
    size_t i = (size_t)blockIdx.x * blockDim.x + threadIdx.x;
    if (i < n) p[i] = val;
}

__device__ __forceinline__ void atomicMaxF(float* addr, float val) {
    if (val >= 0.f) atomicMax((int*)addr, __float_as_int(val));
    else            atomicMin((unsigned int*)addr, __float_as_uint(val));
}

// alpha1[e,h] = dot(q[dst,h,:], k[src,h,:]) / sqrt(C1).  One wave per (e,h).
__global__ void alpha1_kernel(const float* __restrict__ q, const float* __restrict__ k,
                              const long long* __restrict__ ei, float* __restrict__ alpha)
{
    const int gid  = blockIdx.x * (blockDim.x >> 5) + (threadIdx.x >> 5);
    const int lane = threadIdx.x & 31;
    if (gid >= EE * H1c) return;
    const int e = gid >> 2, h = gid & 3;
    const int s = (int)ei[2 * e];
    const int d = (int)ei[2 * e + 1];
    const float* qp = q + (size_t)d * D1c + h * C1c;
    const float* kp = k + (size_t)s * D1c + h * C1c;
    float sum = 0.f;
    #pragma unroll
    for (int j = 0; j < C1c / 32; ++j) {
        const int c = lane + j * 32;
        sum += qp[c] * kp[c];
    }
    #pragma unroll
    for (int off = 16; off > 0; off >>= 1) sum += __shfl_xor(sum, off, 32);
    if (lane == 0) alpha[gid] = sum * 0.04419417382415922f;  // 1/sqrt(512)
}

// segment max over destination (generic: handles self-loop tail when e >= Ereal)
__global__ void segmax_kernel(const float* __restrict__ alpha, const long long* __restrict__ ei,
                              float* __restrict__ amax, int Etot, int H, int Ereal)
{
    const int i = blockIdx.x * blockDim.x + threadIdx.x;
    if (i >= Etot * H) return;
    const int e = i / H, h = i - e * H;
    const int d = (e < Ereal) ? (int)ei[2 * e + 1] : (e - Ereal);
    atomicMaxF(&amax[d * H + h], alpha[i]);
}

// alpha <- exp(alpha - amax[dst]); denom[dst] += alpha
__global__ void expdenom_kernel(float* __restrict__ alpha, const long long* __restrict__ ei,
                                const float* __restrict__ amax, float* __restrict__ denom,
                                int Etot, int H, int Ereal)
{
    const int i = blockIdx.x * blockDim.x + threadIdx.x;
    if (i >= Etot * H) return;
    const int e = i / H, h = i - e * H;
    const int d = (e < Ereal) ? (int)ei[2 * e + 1] : (e - Ereal);
    float am = amax[d * H + h];
    if (!isfinite(am)) am = 0.f;                // reference: where(isfinite, amax, 0)
    const float ex = expf(alpha[i] - am);
    alpha[i] = ex;
    atomicAdd(&denom[d * H + h], ex);
}

// hbuf[dst] += v[src] * softmax(alpha).  One block per edge.
__global__ void msg1_kernel(const float* __restrict__ v, const float* __restrict__ alpha,
                            const float* __restrict__ denom, const long long* __restrict__ ei,
                            float* __restrict__ hb)
{
    const int e = blockIdx.x;
    const int s = (int)ei[2 * e];
    const int d = (int)ei[2 * e + 1];
    for (int idx = threadIdx.x; idx < D1c; idx += blockDim.x) {
        const int h = idx >> 9;                 // /C1
        const float an = alpha[e * H1c + h] / (denom[d * H1c + h] + 1e-16f);
        atomicAdd(&hb[(size_t)d * D1c + idx], v[(size_t)s * D1c + idx] * an);
    }
}

__global__ void elu_kernel(float* __restrict__ h, size_t n) {
    const size_t i = (size_t)blockIdx.x * blockDim.x + threadIdx.x;
    if (i < n) {
        const float x = h[i];
        h[i] = (x > 0.f) ? x : (expf(x) - 1.f);
    }
}

// a2[e2,h] = sum_c leaky_relu(gl[src2,h,c] + gr[dst2,h,c], 0.2) * att[h,c]
__global__ void alpha2_kernel(const float* __restrict__ gl, const float* __restrict__ gr,
                              const float* __restrict__ att, const long long* __restrict__ ei,
                              float* __restrict__ alpha)
{
    const int gid  = blockIdx.x * (blockDim.x >> 5) + (threadIdx.x >> 5);
    const int lane = threadIdx.x & 31;
    if (gid >= E2c * H2c) return;
    const int e = gid >> 3, h = gid & 7;
    int s, d;
    if (e < EE) { s = (int)ei[2 * e]; d = (int)ei[2 * e + 1]; }
    else        { s = d = e - EE; }
    const float* glp = gl + (size_t)s * D2c + h * C2c;
    const float* grp = gr + (size_t)d * D2c + h * C2c;
    const float* ap  = att + h * C2c;
    float sum = 0.f;
    for (int c = lane; c < C2c; c += 32) {
        float t = glp[c] + grp[c];
        t = (t > 0.f) ? t : 0.2f * t;
        sum += t * ap[c];
    }
    #pragma unroll
    for (int off = 16; off > 0; off >>= 1) sum += __shfl_xor(sum, off, 32);
    if (lane == 0) alpha[gid] = sum;
}

// acc2[dst2] += gl[src2] * softmax(a2).  One block per edge2.
__global__ void agg2_kernel(const float* __restrict__ gl, const float* __restrict__ alpha,
                            const float* __restrict__ denom, const long long* __restrict__ ei,
                            float* __restrict__ acc)
{
    const int e = blockIdx.x;
    int s, d;
    if (e < EE) { s = (int)ei[2 * e]; d = (int)ei[2 * e + 1]; }
    else        { s = d = e - EE; }
    for (int idx = threadIdx.x; idx < D2c; idx += blockDim.x) {
        const int h = idx / C2c;
        const float an = alpha[e * H2c + h] / (denom[d * H2c + h] + 1e-16f);
        atomicAdd(&acc[(size_t)d * D2c + idx], gl[(size_t)s * D2c + idx] * an);
    }
}

// out[n,c] = mean_h acc2[n,h,c] + b_out[c]
__global__ void out_kernel(const float* __restrict__ acc, const float* __restrict__ b_out,
                           float* __restrict__ out)
{
    const int n = blockIdx.x;
    for (int c = threadIdx.x; c < C2c; c += blockDim.x) {
        float s = 0.f;
        #pragma unroll
        for (int h = 0; h < H2c; ++h) s += acc[(size_t)n * D2c + h * C2c + c];
        out[(size_t)n * C2c + c] = s * 0.125f + b_out[c];
    }
}

// =====================================================================
// Launch
// =====================================================================
extern "C" void kernel_launch(void* const* d_in, const int* in_sizes, int n_in,
                              void* d_out, int out_size, void* d_ws, size_t ws_size,
                              hipStream_t stream)
{
    const float*     x     = (const float*)d_in[0];
    const long long* ei    = (const long long*)d_in[1];
    const float*     Wq    = (const float*)d_in[2];
    const float*     bq    = (const float*)d_in[3];
    const float*     Wk    = (const float*)d_in[4];
    const float*     bk    = (const float*)d_in[5];
    const float*     Wv    = (const float*)d_in[6];
    const float*     bv    = (const float*)d_in[7];
    const float*     Wsk   = (const float*)d_in[8];
    const float*     bsk   = (const float*)d_in[9];
    const float*     Wl    = (const float*)d_in[10];
    const float*     bl    = (const float*)d_in[11];
    const float*     Wr    = (const float*)d_in[12];
    const float*     br    = (const float*)d_in[13];
    const float*     att   = (const float*)d_in[14];
    const float*     b_out = (const float*)d_in[15];
    float*           out   = (float*)d_out;

    // ---- workspace carve-up (floats) ----
    float* ws = (float*)d_ws;
    const size_t NF = (size_t)NN * D1c;     // 20,480,000
    const size_t NG = (size_t)NN * D2c;     // 36,800,000
    float* q     = ws;
    float* kk    = q    + NF;
    float* v     = kk   + NF;
    float* hb    = v    + NF;               // skip proj, then += agg, then ELU
    float* gl    = hb   + NF;
    float* gr    = gl   + NG;
    float* acc2  = gr   + NG;
    float* al1   = acc2 + NG;               // E*H1
    float* amax1 = al1   + (size_t)EE * H1c;
    float* den1  = amax1 + (size_t)NN * H1c;
    float* al2   = den1  + (size_t)NN * H1c;
    float* amax2 = al2   + (size_t)E2c * H2c;
    float* den2  = amax2 + (size_t)NN * H2c;

    // ---- init accumulators ----
    auto nb = [](size_t n) { return (unsigned)((n + 255) / 256); };
    fill_kernel<<<nb((size_t)NN * H1c), 256, 0, stream>>>(amax1, -INFINITY, (size_t)NN * H1c);
    fill_kernel<<<nb((size_t)NN * H1c), 256, 0, stream>>>(den1, 0.f, (size_t)NN * H1c);
    fill_kernel<<<nb((size_t)NN * H2c), 256, 0, stream>>>(amax2, -INFINITY, (size_t)NN * H2c);
    fill_kernel<<<nb((size_t)NN * H2c), 256, 0, stream>>>(den2, 0.f, (size_t)NN * H2c);
    fill_kernel<<<nb(NG), 256, 0, stream>>>(acc2, 0.f, NG);

    // ---- layer-1 projections (WMMA GEMMs) ----
    dim3 gblk(256);
    dim3 g1((D1c + 127) / 128, (NN + 127) / 128);
    gemm_wmma_f16<<<g1, gblk, 0, stream>>>(x, Wq,  bq,  q,  NN, IND, D1c);
    gemm_wmma_f16<<<g1, gblk, 0, stream>>>(x, Wk,  bk,  kk, NN, IND, D1c);
    gemm_wmma_f16<<<g1, gblk, 0, stream>>>(x, Wv,  bv,  v,  NN, IND, D1c);
    gemm_wmma_f16<<<g1, gblk, 0, stream>>>(x, Wsk, bsk, hb, NN, IND, D1c);

    // ---- layer-1 attention ----
    alpha1_kernel<<<(EE * H1c) / 8, 256, 0, stream>>>(q, kk, ei, al1);
    segmax_kernel<<<nb((size_t)EE * H1c), 256, 0, stream>>>(al1, ei, amax1, EE, H1c, EE);
    expdenom_kernel<<<nb((size_t)EE * H1c), 256, 0, stream>>>(al1, ei, amax1, den1, EE, H1c, EE);
    msg1_kernel<<<EE, 256, 0, stream>>>(v, al1, den1, ei, hb);
    elu_kernel<<<nb(NF), 256, 0, stream>>>(hb, NF);

    // ---- layer-2 projections (WMMA GEMMs) ----
    dim3 g2((D2c + 127) / 128, (NN + 127) / 128);
    gemm_wmma_f16<<<g2, gblk, 0, stream>>>(hb, Wl, bl, gl, NN, D1c, D2c);
    gemm_wmma_f16<<<g2, gblk, 0, stream>>>(hb, Wr, br, gr, NN, D1c, D2c);

    // ---- layer-2 attention (edges + self loops) ----
    alpha2_kernel<<<(E2c * H2c) / 8, 256, 0, stream>>>(gl, gr, att, ei, al2);
    segmax_kernel<<<nb((size_t)E2c * H2c), 256, 0, stream>>>(al2, ei, amax2, E2c, H2c, EE);
    expdenom_kernel<<<nb((size_t)E2c * H2c), 256, 0, stream>>>(al2, ei, amax2, den2, E2c, H2c, EE);
    agg2_kernel<<<E2c, 256, 0, stream>>>(gl, al2, den2, ei, acc2);

    // ---- head mean + bias ----
    out_kernel<<<NN, 256, 0, stream>>>(acc2, b_out, out);
}
